// QuantumGate_65481071407791
// MI455X (gfx1250) — compile-verified, matrix operand order checked by reading, not run on hardware
//
#include <hip/hip_runtime.h>
#include <math.h>

typedef __attribute__((ext_vector_type(2)))  float    v2f;
typedef __attribute__((ext_vector_type(8)))  float    v8f;
typedef __attribute__((ext_vector_type(16))) _Float16 v16h;

#define NQ    12
#define DEPTH 4
#define DIM   4096      // 2^12 amplitudes
#define KDIM  512
#define NPAD  16        // params row stride (12 used, padded to WMMA N=16)

// ---------------------------------------------------------------------------
// Kernel 1: params[M][0..11] = x[M][:] @ W.T + b  via WMMA.
// One wave handles one 16-row M-tile; block = 256 threads = 8 waves = 128 rows.
// f32 path: V_WMMA_F32_16X16X4_F32 (exact f32 accumulation, K-step 4).
//   A (16x4 f32, 2 VGPR): lane<16 -> M=lane, K={k0,k0+1}; lane>=16 -> K={k0+2,k0+3}
//   B (4x16 f32, 2 VGPR): lane<16 -> N=lane, K={k0,k0+1}; lane>=16 -> K={k0+2,k0+3}
//   C (16x16 f32, 8 VGPR): vgpr r, lane<16 -> M=r,N=lane; lane>=16 -> M=r+8,N=lane-16
// ---------------------------------------------------------------------------
__global__ __launch_bounds__(256)
void gemm_params_wmma(const float* __restrict__ x,
                      const float* __restrict__ Wm,
                      const float* __restrict__ bias,
                      float* __restrict__ params) {
    const int lane = threadIdx.x & 31;
    const int wave = threadIdx.x >> 5;
    const int row0 = (blockIdx.x * 8 + wave) * 16;
    const int r    = lane & 15;        // M (A) / N (B) within tile
    const int hi   = lane >> 4;        // K half-select
    const int wrow = (r < NQ) ? r : 0; // clamp: W has only 12 rows (pad cols junk, unused)

    v8f acc = {};

#if __has_builtin(__builtin_amdgcn_wmma_f32_16x16x4_f32)
    {
        const int koff = hi * 2;
        const float* arow = x  + (size_t)(row0 + r) * KDIM + koff;
        const float* brow = Wm + (size_t)wrow       * KDIM + koff;
        for (int k0 = 0; k0 < KDIM; k0 += 4) {
            v2f a  = { arow[k0], arow[k0 + 1] };
            v2f bb = { brow[k0], brow[k0 + 1] };
            acc = __builtin_amdgcn_wmma_f32_16x16x4_f32(
                false, a, false, bb, (short)0, acc, false, false);
        }
    }
#else
    {
        // Fallback: codegen-confirmed f16 WMMA, K-step 32.
        // A 16x32 f16: half h of lane -> K = k0 + (h&7) + 16*(h>>3) + 8*(lane>>4)
        const int koffl = hi * 8;
        const float* arow = x  + (size_t)(row0 + r) * KDIM;
        const float* brow = Wm + (size_t)wrow       * KDIM;
        for (int k0 = 0; k0 < KDIM; k0 += 32) {
            v16h a, bb;
            #pragma unroll
            for (int h = 0; h < 16; ++h) {
                int kk = k0 + (h & 7) + ((h >> 3) * 16) + koffl;
                a[h]  = (_Float16)arow[kk];
                bb[h] = (_Float16)brow[kk];
            }
            acc = __builtin_amdgcn_wmma_f32_16x16x32_f16(
                false, a, false, bb, (short)0, acc, false, false);
        }
    }
#endif

    const float bn = (r < NQ) ? bias[r] : 0.0f;
    #pragma unroll
    for (int rr = 0; rr < 8; ++rr) {
        const int M = row0 + rr + hi * 8;
        params[(size_t)M * NPAD + r] = acc[rr] + bn;
    }
}

// ---------------------------------------------------------------------------
// Kernel 2: 12-qubit statevector simulation, one workgroup per batch element.
// State (re/im split, 32 KB) lives in LDS; gates are LDS-resident passes.
// wire w acts on bit (11-w); wire 0 = MSB (bit 11).
// ---------------------------------------------------------------------------
__global__ __launch_bounds__(128)
void simulate_qc(const float* __restrict__ params, float* __restrict__ out) {
    __shared__ float re[DIM];
    __shared__ float im[DIM];
    __shared__ float red[128];

    const int tid = threadIdx.x;
    const float* th = params + (size_t)blockIdx.x * NPAD;

    float c[NQ], s[NQ];
    #pragma unroll
    for (int w = 0; w < NQ; ++w) {
        const float h = 0.5f * th[w];
        c[w] = cosf(h);
        s[w] = sinf(h);
    }

    for (int i = tid; i < DIM; i += 128) { re[i] = 0.0f; im[i] = 0.0f; }
    __syncthreads();
    if (tid == 0) re[0] = 1.0f;
    __syncthreads();

    for (int d = 0; d < DEPTH; ++d) {
        // RX(theta_w) on each wire
        for (int w = 0; w < NQ; ++w) {
            const int bit  = 11 - w;
            const int mlow = (1 << bit) - 1;
            const float cw = c[w], sw = s[w];
            for (int p = tid; p < DIM / 2; p += 128) {
                const int i0 = ((p >> bit) << (bit + 1)) | (p & mlow);
                const int i1 = i0 | (1 << bit);
                const float x0 = re[i0], y0 = im[i0];
                const float x1 = re[i1], y1 = im[i1];
                re[i0] = cw * x0 + sw * y1;
                im[i0] = cw * y0 - sw * x1;
                re[i1] = cw * x1 + sw * y0;
                im[i1] = cw * y1 - sw * x0;
            }
            __syncthreads();
        }
        // CNOT ladder: control w (bit bt+1), target w+1 (bit bt)
        for (int w = 0; w < NQ - 1; ++w) {
            const int bt   = 10 - w;
            const int mlow = (1 << bt) - 1;
            for (int q = tid; q < DIM / 4; q += 128) {
                const int lo = q & mlow;
                const int hh = q >> bt;
                const int i0 = (hh << (bt + 2)) | (1 << (bt + 1)) | lo; // ctrl=1,tgt=0
                const int i1 = i0 | (1 << bt);                          // ctrl=1,tgt=1
                float t;
                t = re[i0]; re[i0] = re[i1]; re[i1] = t;
                t = im[i0]; im[i0] = im[i1]; im[i1] = t;
            }
            __syncthreads();
        }
    }

    // <Z_0>: + for bit11==0, - for bit11==1
    float local = 0.0f;
    for (int i = tid; i < DIM; i += 128) {
        const float p = re[i] * re[i] + im[i] * im[i];
        local += ((i >> 11) & 1) ? -p : p;
    }
    red[tid] = local;
    __syncthreads();
    #pragma unroll
    for (int off = 64; off > 0; off >>= 1) {
        if (tid < off) red[tid] += red[tid + off];
        __syncthreads();
    }
    if (tid == 0) out[blockIdx.x] = red[0];
}

// ---------------------------------------------------------------------------
extern "C" void kernel_launch(void* const* d_in, const int* in_sizes, int n_in,
                              void* d_out, int out_size, void* d_ws, size_t ws_size,
                              hipStream_t stream) {
    const float* x    = (const float*)d_in[0];  // [B,512]
    const float* Wm   = (const float*)d_in[1];  // [12,512]
    const float* bias = (const float*)d_in[2];  // [12]
    float* out    = (float*)d_out;              // [B]
    float* params = (float*)d_ws;               // [B,16] padded

    const int B = in_sizes[0] / KDIM;           // 4096

    // 8 waves/block, one 16-row tile per wave -> 128 rows/block
    gemm_params_wmma<<<B / 128, 256, 0, stream>>>(x, Wm, bias, params);
    simulate_qc<<<B, 128, 0, stream>>>(params, out);
}
